// LinearAttention_43808666419724
// MI455X (gfx1250) — matrix operand order repaired and checked by reading, not compiled, not run on hardware
//
#include <hip/hip_runtime.h>
#include <hip/hip_bf16.h>
#include <stdint.h>

typedef __bf16 bf16_t;
typedef __attribute__((ext_vector_type(4)))  __bf16 v4bf;
typedef __attribute__((ext_vector_type(8)))  __bf16 v8bf;
typedef __attribute__((ext_vector_type(16))) __bf16 v16bf;
typedef __attribute__((ext_vector_type(8)))  float  v8f;

#define BM 128
#define BN 128
#define BK 32
#define LDT 40   // padded LDS row (bf16): 80B, multiple of 16B, stride 20 dwords spreads banks

__device__ __forceinline__ float elu1(float x) {
  return x > 0.0f ? x + 1.0f : __expf(x);   // elu(x)+1
}

// ---- CDNA5 async global->LDS copy (ASYNCcnt-tracked, no VGPR round-trip) ----
__device__ __forceinline__ void async_load_b128(uint32_t lds_addr, const void* gaddr) {
  asm volatile("global_load_async_to_lds_b128 %0, %1, off"
               :: "v"(lds_addr), "v"(gaddr)
               : "memory");
}
template <int N>
__device__ __forceinline__ void wait_asynccnt() {
  asm volatile("s_wait_asynccnt %0" :: "i"(N) : "memory");
}

// ---------------- elementwise f32 -> bf16 ----------------
__global__ void cvt_f32_to_bf16(const float* __restrict__ in, bf16_t* __restrict__ out, long n) {
  long i = ((long)blockIdx.x * blockDim.x + threadIdx.x) * 4;
  if (i + 3 < n) {
    float4 v = *(const float4*)(in + i);
    v4bf r;
    r[0] = (bf16_t)v.x; r[1] = (bf16_t)v.y; r[2] = (bf16_t)v.z; r[3] = (bf16_t)v.w;
    *(v4bf*)(out + i) = r;
  }
}

// ---------------- ksum[b,e] = sum_s K[b,s,e] ----------------
__global__ void col_sum(const bf16_t* __restrict__ Kb, float* __restrict__ ks, int S, int E) {
  int b = blockIdx.y;
  int e = blockIdx.x * blockDim.x + threadIdx.x;
  const bf16_t* p = Kb + (long)b * S * E + e;
  float acc = 0.f;
  for (int s = 0; s < S; ++s) acc += (float)p[(long)s * E];
  ks[(long)b * E + e] = acc;
}

// ---------------- z[b,i] = 1/(sum_j ksum[b,j]*Q[b,j,i] + eps) ----------------
__global__ void z_kernel(const bf16_t* __restrict__ Qb, const float* __restrict__ ks,
                         float* __restrict__ z, int S, int E) {
  int b = blockIdx.y;
  int i = blockIdx.x * blockDim.x + threadIdx.x;
  const bf16_t* q = Qb + (long)b * S * E + i;
  const float* kb = ks + (long)b * S;
  float acc = 0.f;
  for (int j = 0; j < S; ++j) acc += kb[j] * (float)q[(long)j * E];
  z[(long)b * S + i] = 1.0f / (acc + 1e-6f);
}

// ---------------- generic NT GEMM: C = A[M,K] * B[N,K]^T ----------------
enum { EPI_QKV = 0, EPI_BF16 = 1, EPI_OUT = 2 };

template <int EPI>
__global__ __launch_bounds__(256)
void gemm_nt(const bf16_t* __restrict__ A, long sA,
             const bf16_t* __restrict__ Bm, long sB,
             int M, int N, int K,
             bf16_t* __restrict__ o0, bf16_t* __restrict__ o1, bf16_t* __restrict__ o2,
             float* __restrict__ of, const float* __restrict__ zv,
             const float* __restrict__ bias, long sC)
{
  // 3-deep LDS pipeline: 3 x (10KB + 10KB) = 60KB
  __shared__ bf16_t As[3][BM * LDT];
  __shared__ bf16_t Bs[3][BN * LDT];

  const int tid  = threadIdx.x;
  const int bz   = blockIdx.z;
  const bf16_t* Ab = A  + (long)bz * sA + (long)blockIdx.y * BM * K;
  const bf16_t* Bb = Bm + (long)bz * sB + (long)blockIdx.x * BN * K;

  const int lane = tid & 31;
  const int wave = tid >> 5;
  const int wm   = wave & 3;     // 4 x 32-row groups
  const int wn   = wave >> 2;    // 2 x 64-col groups
  const int hl   = lane >> 4;    // half-wave (0/1)
  const int r    = lane & 15;

  v8f acc[2][4];
  const v8f zero8 = {0.f, 0.f, 0.f, 0.f, 0.f, 0.f, 0.f, 0.f};
#pragma unroll
  for (int i = 0; i < 2; ++i)
#pragma unroll
    for (int j = 0; j < 4; ++j) acc[i][j] = zero8;

  // per stage: each thread issues 2 A-chunks + 2 B-chunks (16B each), async to LDS
  auto issue_stage = [&](int kt, int buf) {
#pragma unroll
    for (int it = 0; it < 2; ++it) {
      int c = tid + it * 256;                 // 512 chunks: 128 rows x 4 chunks
      int row = c >> 2, col8 = (c & 3) << 3;
      async_load_b128((uint32_t)(uintptr_t)&As[buf][row * LDT + col8],
                      Ab + (long)row * K + kt + col8);
    }
#pragma unroll
    for (int it = 0; it < 2; ++it) {
      int c = tid + it * 256;
      int row = c >> 2, col8 = (c & 3) << 3;
      async_load_b128((uint32_t)(uintptr_t)&Bs[buf][row * LDT + col8],
                      Bb + (long)row * K + kt + col8);
    }
  };

  auto compute = [&](int buf) {
    v16bf af[2], bfr[4];
    // A fragment (16x32 bf16): half h holds K = h*8..+7 and 16+h*8..+7, row = r
#pragma unroll
    for (int i = 0; i < 2; ++i) {
      const bf16_t* p = &As[buf][(wm * 32 + i * 16 + r) * LDT + hl * 8];
      v8bf lo = *(const v8bf*)p;
      v8bf hi = *(const v8bf*)(p + 16);
      af[i] = __builtin_shufflevector(lo, hi, 0,1,2,3,4,5,6,7,8,9,10,11,12,13,14,15);
    }
    // B fragment (32x16 bf16 from [N,K] rows): half h holds K = h*8..+7 and 16+h*8..+7, col = r
#pragma unroll
    for (int j = 0; j < 4; ++j) {
      const bf16_t* p = &Bs[buf][(wn * 64 + j * 16 + r) * LDT + hl * 8];
      v8bf lo = *(const v8bf*)p;
      v8bf hi = *(const v8bf*)(p + 16);
      bfr[j] = __builtin_shufflevector(lo, hi, 0,1,2,3,4,5,6,7,8,9,10,11,12,13,14,15);
    }
#pragma unroll
    for (int i = 0; i < 2; ++i)
#pragma unroll
      for (int j = 0; j < 4; ++j)
        acc[i][j] = __builtin_amdgcn_wmma_f32_16x16x32_bf16(
            false, af[i], false, bfr[j], (short)0, acc[i][j], false, false);
  };

  const int nk = K / BK;                 // >= 2 for all call sites (K = 1024)
  // prologue: 2 stages in flight
  issue_stage(0, 0);
  issue_stage(BK, 1);

  int b0 = 0, b1 = 1, b2 = 2;            // compute buf / next / issue target
  // steady state: straight-line, 2-deep async prefetch
  for (int t = 0; t < nk - 2; ++t) {
    issue_stage((t + 2) * BK, b2);
    wait_asynccnt<8>();                  // <=2 stages outstanding => stage t resident
    __syncthreads();
    compute(b0);
    __syncthreads();
    int tmp = b0; b0 = b1; b1 = b2; b2 = tmp;
  }
  // drain stage nk-2
  wait_asynccnt<4>();
  __syncthreads();
  compute(b0);
  __syncthreads();
  // drain stage nk-1
  wait_asynccnt<0>();
  __syncthreads();
  compute(b1);

  // ---------------- epilogue ----------------
  const int row0 = blockIdx.y * BM + wm * 32;
  const int col0 = blockIdx.x * BN + wn * 64;
#pragma unroll
  for (int i = 0; i < 2; ++i) {
#pragma unroll
    for (int j = 0; j < 4; ++j) {
#pragma unroll
      for (int v = 0; v < 8; ++v) {
        int rr = row0 + i * 16 + hl * 8 + v;   // C layout: VGPR v -> M = v + 8*half
        int cc = col0 + j * 16 + r;            //           lane&15 -> N
        float val = acc[i][j][v];
        if (EPI == EPI_QKV) {
          int seg = cc >> 10, c2 = cc & 1023;
          bf16_t* dst = (seg == 0) ? o0 : ((seg == 1) ? o1 : o2);
          float fv = (seg < 2) ? elu1(val) : val;   // elu+1 on q,k only
          dst[(long)rr * 1024 + c2] = (bf16_t)fv;
        } else if (EPI == EPI_BF16) {
          o0[(long)bz * sC + (long)rr * N + cc] = (bf16_t)val;
        } else { // EPI_OUT: o = z[row]*acc + bias[col], f32
          of[(long)bz * sC + (long)rr * N + cc] = zv[(long)bz * M + rr] * val + bias[cc];
        }
      }
    }
  }
}

extern "C" void kernel_launch(void* const* d_in, const int* in_sizes, int n_in,
                              void* d_out, int out_size, void* d_ws, size_t ws_size,
                              hipStream_t stream) {
  (void)in_sizes; (void)n_in; (void)out_size; (void)ws_size;
  const float* x     = (const float*)d_in[0];
  const float* W_qkv = (const float*)d_in[1];
  const float* W_o   = (const float*)d_in[2];
  const float* b_o   = (const float*)d_in[3];
  float* out = (float*)d_out;

  const int  Bn = 64, S = 1024, E = 1024, D = 1024;
  const long SE  = (long)S * E;
  const long MBS = (long)Bn * S;   // 65536 rows

  char* ws = (char*)d_ws;
  size_t off = 0;
  auto alloc = [&](size_t bytes) -> void* {
    void* p = ws + off;
    off += (bytes + 255) & ~(size_t)255;
    return p;
  };
  bf16_t* Xb   = (bf16_t*)alloc((size_t)MBS * D * sizeof(bf16_t));
  bf16_t* Wqb  = (bf16_t*)alloc((size_t)3 * E * D * sizeof(bf16_t));
  bf16_t* Wob  = (bf16_t*)alloc((size_t)E * E * sizeof(bf16_t));
  bf16_t* Qb   = (bf16_t*)alloc((size_t)Bn * SE * sizeof(bf16_t));
  bf16_t* Kb   = (bf16_t*)alloc((size_t)Bn * SE * sizeof(bf16_t));
  bf16_t* Vb   = (bf16_t*)alloc((size_t)Bn * SE * sizeof(bf16_t));
  bf16_t* kvT  = (bf16_t*)alloc((size_t)Bn * SE * sizeof(bf16_t));
  bf16_t* numb = (bf16_t*)alloc((size_t)Bn * SE * sizeof(bf16_t));
  float*  ksum = (float*)alloc((size_t)Bn * S * sizeof(float));
  float*  zv   = (float*)alloc((size_t)Bn * S * sizeof(float));

  long nX = MBS * D;
  cvt_f32_to_bf16<<<dim3((unsigned)(nX / 1024)), 256, 0, stream>>>(x, Xb, nX);
  long nWq = (long)3 * E * D;
  cvt_f32_to_bf16<<<dim3((unsigned)(nWq / 1024)), 256, 0, stream>>>(W_qkv, Wqb, nWq);
  long nWo = (long)E * E;
  cvt_f32_to_bf16<<<dim3((unsigned)(nWo / 1024)), 256, 0, stream>>>(W_o, Wob, nWo);

  // 1) qkv = X @ Wqkv^T ; elu+1 on q,k ; split into Qb,Kb,Vb (bf16)
  gemm_nt<EPI_QKV><<<dim3(3 * E / BN, (unsigned)(MBS / BM), 1), 256, 0, stream>>>(
      Xb, 0, Wqb, 0, (int)MBS, 3 * E, D, Qb, Kb, Vb, nullptr, nullptr, nullptr, 0);

  // 2) ksum, z
  col_sum<<<dim3(E / 256, Bn), 256, 0, stream>>>(Kb, ksum, S, E);
  z_kernel<<<dim3(E / 256, Bn), 256, 0, stream>>>(Qb, ksum, zv, S, E);

  // 3) kvT[b] = V[b] @ K[b]^T   (kv transposed so the next GEMM is NT again)
  gemm_nt<EPI_BF16><<<dim3(S / BN, S / BM, Bn), 256, 0, stream>>>(
      Vb, SE, Kb, SE, S, S, E, kvT, nullptr, nullptr, nullptr, nullptr, nullptr, SE);

  // 4) num[b] = Q[b] @ kvT[b]^T
  gemm_nt<EPI_BF16><<<dim3(S / BN, S / BM, Bn), 256, 0, stream>>>(
      Qb, SE, kvT, SE, S, S, E, numb, nullptr, nullptr, nullptr, nullptr, nullptr, SE);

  // 5) o[b] = diag(z[b]) * num[b] @ W_o^T + b_o   (f32 out)
  gemm_nt<EPI_OUT><<<dim3(E / BN, S / BM, Bn), 256, 0, stream>>>(
      numb, SE, Wob, 0, S, E, S, nullptr, nullptr, nullptr, out, zv, b_o, SE);
}
